// PodDeepONet2_42245298323675
// MI455X (gfx1250) — matrix operand order repaired and checked by reading, not compiled, not run on hardware
//
#include <hip/hip_runtime.h>

typedef __attribute__((ext_vector_type(16))) _Float16 v16h;
typedef __attribute__((ext_vector_type(8)))  _Float16 v8h;
typedef __attribute__((ext_vector_type(8)))  float    v8f;

#define NN 256        // noise samples
#define MM 32768      // coordinate points
#define PP 128        // modes per field
#define NMF 8388608   // NN*MM

struct MlpP  { const float* W[5]; const float* b[5]; };
struct MlpP3 { MlpP p[3]; };

union HV16 { v16h v; v8h h[2]; };

__device__ inline v8f wmma_f16(v16h a, v16h b, v8f c) {
  return __builtin_amdgcn_wmma_f32_16x16x32_f16(
      /*neg_a=*/false, a, /*neg_b=*/false, b,
      /*c_mod=*/(short)0, c, /*reuse_a=*/false, /*reuse_b=*/false);
}

// A fragment (16x32 f16, rows row0..row0+15, K k0..k0+31) from row-major [row][ld]
// ISA layout: lanes0-15 -> elems 0..7 = K(0..7), 8..15 = K(16..23); lanes16-31 shifted +8.
__device__ inline v16h frag_a(const _Float16* src, int ld, int row0, int k0, int lane) {
  int hf = (lane >> 4) & 1;
  int mr = lane & 15;
  const _Float16* base = src + (row0 + mr) * ld + k0 + hf * 8;
  HV16 u;
  u.h[0] = *(const v8h*)(base);
  u.h[1] = *(const v8h*)(base + 16);
  return u.v;
}

// B fragment (32x16 f16, K k0..k0+31, N n0..n0+15) from K-contiguous transposed [n][ld]
// ISA layout: lane = N col (lane&15); elems 0..7 = K(half*8 + 0..7), 8..15 = K(16 + half*8 + 0..7).
__device__ inline v16h frag_b(const _Float16* srcT, int ld, int n0, int k0, int lane) {
  int hf = (lane >> 4) & 1;
  int nc = lane & 15;
  const _Float16* base = srcT + (n0 + nc) * ld + k0 + hf * 8;
  HV16 u;
  u.h[0] = *(const v8h*)(base);
  u.h[1] = *(const v8h*)(base + 16);
  return u.v;
}

__device__ inline float fast_tanh(float x) {
  float e = __expf(2.0f * x);          // TRANS unit
  return 1.0f - 2.0f / (e + 1.0f);
}

// ---------------------------------------------------------------------------
// Branch MLPs: 3 fields x (256 x [64->32->32->32->32->128]), tanh hidden.
// Tiny: one thread per noise row, scalar math, f16 output for WMMA consumption.
// ---------------------------------------------------------------------------
__global__ __launch_bounds__(256) void branch_kernel(const float* __restrict__ noise,
                                                     MlpP3 PP3,
                                                     _Float16* __restrict__ outB) {
  const MlpP P = PP3.p[blockIdx.x];
  const int row = threadIdx.x;

  float x[64];
  #pragma unroll
  for (int i = 0; i < 64; ++i) x[i] = noise[row * 64 + i];

  float h[32];
  // layer 0: 64 -> 32, tanh
  #pragma unroll 4
  for (int j = 0; j < 32; ++j) {
    float s = P.b[0][j];
    #pragma unroll
    for (int i = 0; i < 64; ++i) s += x[i] * P.W[0][i * 32 + j];
    h[j] = fast_tanh(s);
  }
  // layers 1..3: 32 -> 32, tanh
  for (int l = 1; l <= 3; ++l) {
    float g[32];
    #pragma unroll 4
    for (int j = 0; j < 32; ++j) {
      float s = P.b[l][j];
      #pragma unroll
      for (int i = 0; i < 32; ++i) s += h[i] * P.W[l][i * 32 + j];
      g[j] = fast_tanh(s);
    }
    #pragma unroll
    for (int j = 0; j < 32; ++j) h[j] = g[j];
  }
  // layer 4: 32 -> 128, linear, store f16
  _Float16* o = outB + (size_t)blockIdx.x * NN * PP + (size_t)row * PP;
  #pragma unroll 4
  for (int j = 0; j < 128; ++j) {
    float s = P.b[4][j];
    #pragma unroll
    for (int i = 0; i < 32; ++i) s += h[i] * P.W[4][i * 128 + j];
    o[j] = (_Float16)s;
  }
}

// ---------------------------------------------------------------------------
// Trunk MLP: 32768 x [2->128->128->128->128->outdim], tanh hidden.
// 256 blocks x 128 rows, 8 waves (16 rows each). Activations in LDS (f16,
// padded stride). Weights transposed f32->f16 into LDS so B-fragments are
// contiguous per-lane loads. 4x v_wmma_f32_16x16x32_f16 per 16x16 tile.
// Next layer's weights are prefetched (global_prefetch_b8) during compute.
// means==0: outdim=384, f16 -> outT.   means==1: outdim=3, scalar -> biasOut.
// ---------------------------------------------------------------------------
#define LDA 136  // 128 + 8 halves pad (keeps 16B alignment, spreads banks)

__global__ __launch_bounds__(256) void trunk_kernel(const float* __restrict__ coords,
                                                    MlpP P,
                                                    _Float16* __restrict__ outT,
                                                    float* __restrict__ biasOut,
                                                    int means) {
  __shared__ _Float16 act[128 * LDA];
  __shared__ _Float16 wt[128 * LDA];
  __shared__ float    bsh[128];

  const int tid  = threadIdx.x;
  const int lane = tid & 31;
  const int wave = tid >> 5;
  const int row0 = wave * 16;
  const int m0   = blockIdx.x * 128;
  const int hf   = lane >> 4;
  const int nc   = lane & 15;

  // prefetch layer-1 weights while layer 0 runs (256 thr x 256B = 64KB)
  __builtin_prefetch(P.W[1] + tid * 64, 0, 0);

  // layer 0: 2 -> 128 (scalar, trivially small K)
  for (int idx = tid; idx < 128 * 128; idx += 256) {
    int r = idx >> 7, c = idx & 127;
    float x0 = coords[(size_t)(m0 + r) * 2 + 0];
    float x1 = coords[(size_t)(m0 + r) * 2 + 1];
    float s = P.b[0][c] + x0 * P.W[0][c] + x1 * P.W[0][128 + c];
    act[r * LDA + c] = (_Float16)fast_tanh(s);
  }

  // layers 1..3: 128 -> 128, tanh, WMMA
  for (int l = 1; l <= 3; ++l) {
    __syncthreads();  // all reads of previous wt done
    const float* W = P.W[l];
    for (int idx = tid; idx < 128 * 128; idx += 256) {
      int k = idx >> 7, n = idx & 127;          // coalesced global read
      wt[n * LDA + k] = (_Float16)W[k * 128 + n];  // transposed LDS store
    }
    if (tid < 128) bsh[tid] = P.b[l][tid];
    // prefetch next layer's weights behind this layer's compute
    if (l < 3) __builtin_prefetch(P.W[l + 1] + tid * 64, 0, 0);
    else       __builtin_prefetch(P.W[4] + tid * 64, 0, 0);
    __syncthreads();

    // own-row A fragments (wave-private rows -> safe to overwrite after)
    v16h a0 = frag_a(act, LDA, row0,  0, lane);
    v16h a1 = frag_a(act, LDA, row0, 32, lane);
    v16h a2 = frag_a(act, LDA, row0, 64, lane);
    v16h a3 = frag_a(act, LDA, row0, 96, lane);

    #pragma unroll
    for (int nt = 0; nt < 8; ++nt) {
      v8f acc = {};
      acc = wmma_f16(a0, frag_b(wt, LDA, nt * 16,  0, lane), acc);
      acc = wmma_f16(a1, frag_b(wt, LDA, nt * 16, 32, lane), acc);
      acc = wmma_f16(a2, frag_b(wt, LDA, nt * 16, 64, lane), acc);
      acc = wmma_f16(a3, frag_b(wt, LDA, nt * 16, 96, lane), acc);
      int col = nt * 16 + nc;
      float bv = bsh[col];
      #pragma unroll
      for (int r = 0; r < 8; ++r) {   // D layout: M = r + half*8, N = lane&15
        int row = row0 + r + hf * 8;
        act[row * LDA + col] = (_Float16)fast_tanh(acc[r] + bv);
      }
    }
  }

  if (means == 0) {
    // layer 4: 128 -> 384 linear, f16 out, processed in 3 chunks of 128 cols
    const float* W = P.W[4];
    v16h a0 = frag_a(act, LDA, row0,  0, lane);
    v16h a1 = frag_a(act, LDA, row0, 32, lane);
    v16h a2 = frag_a(act, LDA, row0, 64, lane);
    v16h a3 = frag_a(act, LDA, row0, 96, lane);
    for (int c = 0; c < 3; ++c) {
      __syncthreads();
      for (int idx = tid; idx < 128 * 128; idx += 256) {
        int k = idx >> 7, n = idx & 127;
        wt[n * LDA + k] = (_Float16)W[k * 384 + c * 128 + n];
      }
      if (tid < 128) bsh[tid] = P.b[4][c * 128 + tid];
      __syncthreads();
      #pragma unroll
      for (int nt = 0; nt < 8; ++nt) {
        v8f acc = {};
        acc = wmma_f16(a0, frag_b(wt, LDA, nt * 16,  0, lane), acc);
        acc = wmma_f16(a1, frag_b(wt, LDA, nt * 16, 32, lane), acc);
        acc = wmma_f16(a2, frag_b(wt, LDA, nt * 16, 64, lane), acc);
        acc = wmma_f16(a3, frag_b(wt, LDA, nt * 16, 96, lane), acc);
        int col = nt * 16 + nc;
        float bv = bsh[col];
        #pragma unroll
        for (int r = 0; r < 8; ++r) {
          int row = m0 + row0 + r + hf * 8;
          outT[(size_t)row * 384 + c * 128 + col] = (_Float16)(acc[r] + bv);
        }
      }
    }
  } else {
    // layer 4: 128 -> 3 linear, f32 bias buffer (scalar; only 384 outputs/block)
    __syncthreads();
    const float* W = P.W[4];
    for (int idx = tid; idx < 128 * 3; idx += 256) {
      int r = idx / 3, j = idx % 3;
      float s = P.b[4][j];
      for (int k = 0; k < 128; ++k) s += (float)act[r * LDA + k] * W[k * 3 + j];
      biasOut[(size_t)(m0 + r) * 3 + j] = s;
    }
  }
}

// ---------------------------------------------------------------------------
// Fused final GEMM + bias: for field f, C(n,m) = outB_f(n,:) . outT(m, f*128:)
// + bias(m,f). Grid (256 m-tiles, 3 fields), 16 waves; each wave owns a 16-row
// n-strip and the full 128-col m-tile. outT tile staged into LDS with
// GLOBAL_LOAD_ASYNC_TO_LDS_B128 (ASYNCcnt), results streamed with NT stores.
// outT (25MB f16) is L2-resident (192MB L2), so B-tiles read at L2 bandwidth;
// HBM traffic ~= the 100MB of f32 output stores.
// ---------------------------------------------------------------------------
__global__ __launch_bounds__(512) void fuse_kernel(const _Float16* __restrict__ outB,
                                                   const _Float16* __restrict__ outT,
                                                   const float* __restrict__ biasBuf,
                                                   float* __restrict__ out) {
  __shared__ _Float16 bt[128 * LDA];
  __shared__ float    bsh[128];

  const int f   = blockIdx.y;
  const int m0  = blockIdx.x * 128;
  const int tid = threadIdx.x;
  const int lane = tid & 31;
  const int wave = tid >> 5;
  const int n0  = wave * 16;   // 16 waves x 16 rows = 256 = N
  const int hf  = lane >> 4;
  const int nc  = lane & 15;

  // async-stage B tile: bt[m_local][k] = outT[m0+m][f*128 + k] (K contiguous).
  // 128 rows x 256B = 2048 16B chunks; 512 threads x 4 iterations.
  {
    const unsigned ldsBase = (unsigned)(uintptr_t)(void*)bt;
    const _Float16* gbase = outT + (size_t)m0 * 384 + (size_t)f * 128;
    #pragma unroll
    for (int it = 0; it < 4; ++it) {
      int idx = tid + it * 512;
      int m = idx >> 4, kc = idx & 15;                  // kc: 16B chunk in row
      unsigned lds = ldsBase + (unsigned)(m * LDA + kc * 8) * 2u;
      unsigned long long ga =
          (unsigned long long)(uintptr_t)(gbase + (size_t)m * 384 + kc * 8);
      asm volatile("global_load_async_to_lds_b128 %0, %1, off"
                   :: "v"(lds), "v"(ga) : "memory");
    }
  }
  if (tid < 128) bsh[tid] = biasBuf[(size_t)(m0 + tid) * 3 + f];

  // A fragments straight from global (outB is tiny / L2-resident)
  const _Float16* Abase = outB + (size_t)f * NN * PP;
  v16h a0 = frag_a(Abase, PP, n0,  0, lane);
  v16h a1 = frag_a(Abase, PP, n0, 32, lane);
  v16h a2 = frag_a(Abase, PP, n0, 64, lane);
  v16h a3 = frag_a(Abase, PP, n0, 96, lane);

  asm volatile("s_wait_asynccnt 0x0" ::: "memory");  // own async copies done
  __syncthreads();                                   // everyone's copies done

  #pragma unroll
  for (int nt = 0; nt < 8; ++nt) {
    v8f acc = {};
    acc = wmma_f16(a0, frag_b(bt, LDA, nt * 16,  0, lane), acc);
    acc = wmma_f16(a1, frag_b(bt, LDA, nt * 16, 32, lane), acc);
    acc = wmma_f16(a2, frag_b(bt, LDA, nt * 16, 64, lane), acc);
    acc = wmma_f16(a3, frag_b(bt, LDA, nt * 16, 96, lane), acc);
    int m = m0 + nt * 16 + nc;            // WMMA-N == coordinate index m
    float bv = bsh[nt * 16 + nc];
    size_t base = (size_t)f * NMF + (size_t)m;
    #pragma unroll
    for (int r = 0; r < 8; ++r) {          // WMMA-M == noise index n
      size_t n = (size_t)(n0 + r + hf * 8);
      // write-once stream: non-temporal so it doesn't evict outT from L2
      __builtin_nontemporal_store(acc[r] + bv, &out[base + n * MM]);
    }
  }
}

// ---------------------------------------------------------------------------
// Host side
// ---------------------------------------------------------------------------
extern "C" void kernel_launch(void* const* d_in, const int* in_sizes, int n_in,
                              void* d_out, int out_size, void* d_ws, size_t ws_size,
                              hipStream_t stream) {
  (void)in_sizes; (void)n_in; (void)out_size; (void)ws_size;

  const float* noise  = (const float*)d_in[0];
  const float* coords = (const float*)d_in[1];

  auto getP = [&](int base) {
    MlpP p;
    for (int l = 0; l < 5; ++l) {
      p.W[l] = (const float*)d_in[base + 2 * l];
      p.b[l] = (const float*)d_in[base + 2 * l + 1];
    }
    return p;
  };
  MlpP3 PB;
  PB.p[0] = getP(2);    // branch_e
  PB.p[1] = getP(12);   // branch_x
  PB.p[2] = getP(22);   // branch_y
  MlpP Pc = getP(32);   // trunk_comp
  MlpP Pm = getP(42);   // trunk_means

  // workspace layout: outB f16 [3][256][128] | outT f16 [32768][384] | bias f32 [32768][3]
  _Float16* outB = (_Float16*)d_ws;
  _Float16* outT = outB + (size_t)3 * NN * PP;
  float* biasBuf = (float*)(outT + (size_t)MM * 384);
  float* out = (float*)d_out;

  branch_kernel<<<dim3(3), dim3(256), 0, stream>>>(noise, PB, outB);
  trunk_kernel<<<dim3(MM / 128), dim3(256), 0, stream>>>(coords, Pc, outT, biasBuf, 0);
  trunk_kernel<<<dim3(MM / 128), dim3(256), 0, stream>>>(coords, Pm, outT, biasBuf, 1);
  fuse_kernel<<<dim3(MM / 128, 3), dim3(512), 0, stream>>>(outB, outT, biasBuf, out);
}
